// ASTGCN_with_GAT_10419590660155
// MI455X (gfx1250) — matrix operand order repaired and checked by reading, not compiled, not run on hardware
//
#include <hip/hip_runtime.h>
#include <hip/hip_bf16.h>

typedef __attribute__((ext_vector_type(16))) _Float16 v16h;
typedef __attribute__((ext_vector_type(8)))  float    v8f;

#define NEG_SLOPE 0.2f
#define HIDW 256   // hidden width (H*OUT)
#define HEADS 4
#define OUTW 64
#define BK 32      // K step (one WMMA K)
#define BN 64      // cols per block

// ---------------------------------------------------------------------------
// GEMM: C[M x 256] = A[M x K] * B[K x 256], fp32 in/out, f16 WMMA compute.
// Block = 8 waves = 128 rows x 64 cols. Per K-step the 32x64 fp32 B slab is
// DMA'd into LDS with global_load_async_to_lds_b128 (ASYNCcnt), then every
// wave builds its 4 B fragments from LDS and issues 4 WMMAs.
// Fragment layouts per CDNA5 ISA 7.12.2 (wave32):
//   A 16x32 f16 : lane L -> m = L%16 ; elem j: v=j/2,half=j%2 ;
//                 K = (v<4 ? 8*(L/16)+2*v+half : 16+8*(L/16)+2*(v-4)+half)
//   B 32x16 f16 : lane L -> n = L%16 ; elem j -> K = (j<8 ? 8*(L/16)+j : 16+8*(L/16)+j-8)
//   C/D 16x16 f32: VGPR i -> row = i + 8*(L/16), col = L%16
// ---------------------------------------------------------------------------
__global__ void gat_gemm_wmma(const float* __restrict__ A,
                              const float* __restrict__ B,
                              float* __restrict__ C,
                              int M, int K, int Mtiles) {
  __shared__ float bs[BK * BN];                    // 8 KB staging slab [k][n]
  const int lane  = threadIdx.x & 31;
  const int wave  = threadIdx.x >> 5;
  const int mt    = blockIdx.x * 8 + wave;         // 16-row tile per wave
  const bool active = (mt < Mtiles);               // tail waves still barrier
  const int colbase = blockIdx.y * BN;
  const int mrow  = (active ? mt : 0) * 16 + (lane & 15);
  const int khalf = lane >> 4;
  const unsigned ldsbase = (unsigned)(size_t)&bs[0];

  v8f acc[4] = {};
  for (int k0 = 0; k0 < K; k0 += BK) {
    __syncthreads();                               // prev slab fully consumed
    // ---- async DMA: B[k0..k0+31][colbase..colbase+63] -> LDS (2x16B/thread)
#pragma unroll
    for (int c = 0; c < 2; ++c) {
      const int chunk = (int)threadIdx.x + c * 256;          // 0..511
      const int r  = chunk >> 4;                             // k row 0..31
      const int cc = (chunk & 15) << 2;                      // col (floats)
      const unsigned lds_off = ldsbase + (unsigned)(r * BN + cc) * 4u;
      const unsigned long long gaddr =
          (unsigned long long)(size_t)(B + (size_t)(k0 + r) * HIDW + colbase + cc);
      asm volatile("global_load_async_to_lds_b128 %0, %1, off"
                   :: "v"(lds_off), "v"(gaddr) : "memory");
    }
    asm volatile("s_wait_asynccnt 0x0" ::: "memory");        // my chunks done
    __syncthreads();                                         // slab visible

    // ---- A fragment (direct global, rows private to this wave)
    v16h a;
#pragma unroll
    for (int j = 0; j < 16; ++j) {
      const int v = j >> 1, half = j & 1;
      const int kk = (v < 4) ? (8 * khalf + 2 * v + half)
                             : (16 + 8 * khalf + 2 * (v - 4) + half);
      a[j] = (_Float16)A[(size_t)mrow * K + k0 + kk];
    }
    // ---- 4 B fragments from LDS + 4 WMMAs
#pragma unroll
    for (int t = 0; t < 4; ++t) {
      v16h b;
#pragma unroll
      for (int j = 0; j < 16; ++j)
        b[j] = (_Float16)bs[((j < 8) ? (8 * khalf + j) : (16 + 8 * khalf + j - 8)) * BN + t * 16 + (lane & 15)];
      acc[t] = __builtin_amdgcn_wmma_f32_16x16x32_f16(
          false, a, false, b, (short)0, acc[t], false, false);
    }
  }
  if (active) {
#pragma unroll
    for (int t = 0; t < 4; ++t) {
      const int n = colbase + t * 16 + (lane & 15);
#pragma unroll
      for (int i = 0; i < 8; ++i) {
        const int row = mt * 16 + i + 8 * khalf;
        C[(size_t)row * HIDW + n] = acc[t][i];
      }
    }
  }
}

// per-node attention logits: s_src[n,h] = sum_o h[n,h,o]*a_src[h,o]
__global__ void gat_logits(const float* __restrict__ h,
                           const float* __restrict__ a_src,
                           const float* __restrict__ a_dst,
                           float* __restrict__ s_src,
                           float* __restrict__ s_dst, int Nn) {
  const int idx = blockIdx.x * blockDim.x + threadIdx.x;
  if (idx >= Nn * HEADS) return;
  const int n = idx >> 2, head = idx & 3;
  const float* hp = h + (size_t)n * HIDW + head * OUTW;
  const float* as = a_src + head * OUTW;
  const float* ad = a_dst + head * OUTW;
  float ss = 0.f, sd = 0.f;
#pragma unroll 8
  for (int o = 0; o < OUTW; ++o) {
    const float v = hp[o];
    ss = fmaf(v, as[o], ss);
    sd = fmaf(v, ad[o], sd);
  }
  s_src[idx] = ss;
  s_dst[idx] = sd;
}

__device__ __forceinline__ unsigned mono_map(float f) {
  const unsigned b = __float_as_uint(f);
  return (b & 0x80000000u) ? ~b : (b | 0x80000000u);
}
__device__ __forceinline__ float mono_unmap(unsigned u) {
  return (u & 0x80000000u) ? __uint_as_float(u & 0x7fffffffu)
                           : __uint_as_float(~u);
}

// pass 1: e = leaky_relu(s_src[src]+s_dst[dst]); segment-max via u32 atomicMax
__global__ void gat_edge_scores(const int* __restrict__ src_idx,
                                const int* __restrict__ dst_idx,
                                int E_, int Etot,
                                const float* __restrict__ s_src,
                                const float* __restrict__ s_dst,
                                float* __restrict__ e_buf,
                                unsigned* __restrict__ m_u) {
  const int e = blockIdx.x * blockDim.x + threadIdx.x;
  if (e >= Etot) return;
  int s, d;
  if (e < E_) { s = src_idx[e]; d = dst_idx[e]; } else { s = d = e - E_; }
  const float4 a = ((const float4*)s_src)[s];
  const float4 b = ((const float4*)s_dst)[d];
  float v[4] = {a.x + b.x, a.y + b.y, a.z + b.z, a.w + b.w};
  float out[4];
#pragma unroll
  for (int hh = 0; hh < HEADS; ++hh) {
    const float x = v[hh];
    const float lr = x > 0.f ? x : NEG_SLOPE * x;
    out[hh] = lr;
    atomicMax(&m_u[d * HEADS + hh], mono_map(lr));
  }
  ((float4*)e_buf)[e] = make_float4(out[0], out[1], out[2], out[3]);
}

// pass 2: ex = exp(e - m[dst]); segment-sum denominator
__global__ void gat_edge_softmax(const int* __restrict__ dst_idx,
                                 int E_, int Etot,
                                 const unsigned* __restrict__ m_u,
                                 float* __restrict__ e_buf,
                                 float* __restrict__ denom) {
  const int e = blockIdx.x * blockDim.x + threadIdx.x;
  if (e >= Etot) return;
  const int d = (e < E_) ? dst_idx[e] : (e - E_);
  const float4 ev = ((const float4*)e_buf)[e];
  float vv[4] = {ev.x, ev.y, ev.z, ev.w};
  float out[4];
#pragma unroll
  for (int hh = 0; hh < HEADS; ++hh) {
    const float mval = mono_unmap(m_u[d * HEADS + hh]);
    const float ex = __expf(vv[hh] - mval);
    out[hh] = ex;
    atomicAdd(&denom[d * HEADS + hh], ex);
  }
  ((float4*)e_buf)[e] = make_float4(out[0], out[1], out[2], out[3]);
}

// pass 3: one wave per edge; agg[dst] += alpha * h[src] (256 floats/edge)
__global__ void gat_edge_aggregate(const int* __restrict__ src_idx,
                                   const int* __restrict__ dst_idx,
                                   int E_, int Etot,
                                   const float* __restrict__ e_buf,
                                   const float* __restrict__ denom,
                                   const float* __restrict__ h,
                                   float* __restrict__ agg) {
  const int e    = blockIdx.x * (blockDim.x >> 5) + (threadIdx.x >> 5);
  const int lane = threadIdx.x & 31;
  if (e >= Etot) return;
  int s, d;
  if (e < E_) { s = src_idx[e]; d = dst_idx[e]; } else { s = d = e - E_; }
  const float4 ex4 = ((const float4*)e_buf)[e];
  const float4 dn4 = ((const float4*)denom)[d];
  const float alpha[4] = {ex4.x / dn4.x, ex4.y / dn4.y,
                          ex4.z / dn4.z, ex4.w / dn4.w};
  const float* hs = h + (size_t)s * HIDW;
  float* ag = agg + (size_t)d * HIDW;
#pragma unroll
  for (int it = 0; it < 8; ++it) {
    const int o = it * 32 + lane;          // head = o/64 = it/2 (wave-uniform)
    atomicAdd(&ag[o], hs[o] * alpha[it >> 1]);
  }
}

__global__ void gat_bias_relu(const float* __restrict__ agg,
                              const float* __restrict__ b,
                              float* __restrict__ xout, int total) {
  const int i = blockIdx.x * blockDim.x + threadIdx.x;
  if (i >= total) return;
  const float v = agg[i] + b[i & (HIDW - 1)];
  xout[i] = v > 0.f ? v : 0.f;
}

// layer-3 epilogue: relu(agg+b3) + x1, then mean-pool accumulation per graph
__global__ void gat_final_node(const float* __restrict__ agg,
                               const float* __restrict__ b,
                               const float* __restrict__ x1,
                               const int* __restrict__ batch,
                               float* __restrict__ pooled,
                               float* __restrict__ cnt, int Nn) {
  const int i = blockIdx.x * blockDim.x + threadIdx.x;
  if (i >= Nn * HIDW) return;
  const int n = i >> 8, c = i & (HIDW - 1);
  float v = agg[i] + b[c];
  v = (v > 0.f ? v : 0.f) + x1[i];
  const int g = batch[n];
  atomicAdd(&pooled[g * HIDW + c], v);
  if (c == 0) atomicAdd(&cnt[g], 1.0f);
}

__global__ void gat_fc(const float* __restrict__ pooled,
                       const float* __restrict__ cnt,
                       const float* __restrict__ fcW,
                       const float* __restrict__ fcb,
                       float* __restrict__ out, int G_, int C_) {
  const int i = blockIdx.x * blockDim.x + threadIdx.x;
  if (i >= G_ * C_) return;
  const int g = i / C_, c = i % C_;
  const float inv = 1.0f / fmaxf(cnt[g], 1.0f);
  float s = fcb[c];
#pragma unroll 8
  for (int k = 0; k < HIDW; ++k)
    s = fmaf(pooled[g * HIDW + k] * inv, fcW[k * C_ + c], s);
  out[i] = s;
}

// ---------------------------------------------------------------------------
extern "C" void kernel_launch(void* const* d_in, const int* in_sizes, int n_in,
                              void* d_out, int out_size, void* d_ws, size_t ws_size,
                              hipStream_t stream) {
  const float* x  = (const float*)d_in[0];
  const int* eidx = (const int*)d_in[1];
  const int* batch= (const int*)d_in[2];
  const float* W1  = (const float*)d_in[3];
  const float* as1 = (const float*)d_in[4];
  const float* ad1 = (const float*)d_in[5];
  const float* b1  = (const float*)d_in[6];
  const float* W2  = (const float*)d_in[7];
  const float* as2 = (const float*)d_in[8];
  const float* ad2 = (const float*)d_in[9];
  const float* b2  = (const float*)d_in[10];
  const float* W3  = (const float*)d_in[11];
  const float* as3 = (const float*)d_in[12];
  const float* ad3 = (const float*)d_in[13];
  const float* b3  = (const float*)d_in[14];
  const float* fcW = (const float*)d_in[15];
  const float* fcb = (const float*)d_in[16];

  const int IN   = 64;
  const int N    = in_sizes[0] / IN;     // 50000
  const int E    = in_sizes[1] / 2;      // 800000
  const int Etot = E + N;                // self loops appended
  const int C    = in_sizes[16];         // 10
  const int G    = 64;

  const int* src_idx = eidx;
  const int* dst_idx = eidx + E;

  // ---- workspace layout (256B aligned slices) ----
  size_t off = 0;
  char* base = (char*)d_ws;
  auto alloc = [&](size_t bytes) -> void* {
    void* p = base + off;
    off += (bytes + 255) & ~(size_t)255;
    return p;
  };
  float*    h_buf  = (float*)   alloc((size_t)N * HIDW * 4);
  float*    agg    = (float*)   alloc((size_t)N * HIDW * 4);
  float*    x1     = (float*)   alloc((size_t)N * HIDW * 4);
  float*    x2     = (float*)   alloc((size_t)N * HIDW * 4);
  float*    s_src  = (float*)   alloc((size_t)N * HEADS * 4);
  float*    s_dst  = (float*)   alloc((size_t)N * HEADS * 4);
  unsigned* m_u    = (unsigned*)alloc((size_t)N * HEADS * 4);
  float*    denom  = (float*)   alloc((size_t)N * HEADS * 4);
  float*    e_buf  = (float*)   alloc((size_t)Etot * HEADS * 4);
  float*    pooled = (float*)   alloc((size_t)G * HIDW * 4);
  float*    cnt    = (float*)   alloc((size_t)G * 4);

  const int Mtiles = N / 16;                       // 3125
  const dim3 gemmGrid((Mtiles + 7) / 8, HIDW / BN);// 391 x 4 blocks
  const int logBlocks  = (N * HEADS + 255) / 256;
  const int edgBlocks  = (Etot + 255) / 256;
  const int aggBlocks  = (Etot + 7) / 8;           // one wave per edge
  const int nodBlocks  = (N * HIDW + 255) / 256;

  auto run_layer = [&](const float* xin, int K, const float* W,
                       const float* as, const float* ad) {
    hipMemsetAsync(m_u,   0, (size_t)N * HEADS * 4, stream);
    hipMemsetAsync(denom, 0, (size_t)N * HEADS * 4, stream);
    hipMemsetAsync(agg,   0, (size_t)N * HIDW * 4, stream);
    gat_gemm_wmma<<<gemmGrid, 256, 0, stream>>>(xin, W, h_buf, N, K, Mtiles);
    gat_logits<<<logBlocks, 256, 0, stream>>>(h_buf, as, ad, s_src, s_dst, N);
    gat_edge_scores<<<edgBlocks, 256, 0, stream>>>(src_idx, dst_idx, E, Etot,
                                                   s_src, s_dst, e_buf, m_u);
    gat_edge_softmax<<<edgBlocks, 256, 0, stream>>>(dst_idx, E, Etot,
                                                    m_u, e_buf, denom);
    gat_edge_aggregate<<<aggBlocks, 256, 0, stream>>>(src_idx, dst_idx, E, Etot,
                                                      e_buf, denom, h_buf, agg);
  };

  // layer 1: x(N x 64) -> x1
  run_layer(x, IN, W1, as1, ad1);
  gat_bias_relu<<<nodBlocks, 256, 0, stream>>>(agg, b1, x1, N * HIDW);

  // layer 2: x1 -> x2
  run_layer(x1, HIDW, W2, as2, ad2);
  gat_bias_relu<<<nodBlocks, 256, 0, stream>>>(agg, b2, x2, N * HIDW);

  // layer 3: x2 -> (relu(agg+b3) + x1) pooled per graph
  run_layer(x2, HIDW, W3, as3, ad3);
  hipMemsetAsync(pooled, 0, (size_t)G * HIDW * 4, stream);
  hipMemsetAsync(cnt,    0, (size_t)G * 4, stream);
  gat_final_node<<<nodBlocks, 256, 0, stream>>>(agg, b3, x1, batch,
                                                pooled, cnt, N);
  gat_fc<<<(G * C + 255) / 256, 256, 0, stream>>>(pooled, cnt, fcW, fcb,
                                                  (float*)d_out, G, C);
}
